// QGKT_51694226375075
// MI455X (gfx1250) — matrix operand order repaired and checked by reading, not compiled, not run on hardware
//
#include <hip/hip_runtime.h>
#include <hip/hip_bf16.h>

typedef __bf16 bf16;
typedef __attribute__((ext_vector_type(16))) __bf16 v16bf;
typedef __attribute__((ext_vector_type(8)))  __bf16 v8bf;
typedef __attribute__((ext_vector_type(8)))  float  v8f;
typedef int v4i_ __attribute__((__vector_size__(16)));
typedef int v8i_ __attribute__((__vector_size__(32)));

union ABfrag { v16bf v; v8bf h[2]; };

static constexpr int Bsz = 64;
static constexpr int T   = 200;
static constexpr int H   = 256;
static constexpr int G3  = 768;     // 3*H
static constexpr long long NX = (long long)Bsz * T * H;   // x elements (time-major rows)
static constexpr long long NW = (long long)G3 * H;        // one weight matrix

#if defined(__HIP_DEVICE_COMPILE__) &&                                        \
    __has_builtin(__builtin_amdgcn_global_load_async_to_lds_b128) &&          \
    __has_builtin(__builtin_amdgcn_s_wait_asynccnt)
#define HAVE_ASYNC_LDS 1
#else
#define HAVE_ASYNC_LDS 0
#endif

#if defined(__HIP_DEVICE_COMPILE__) &&                                        \
    __has_builtin(__builtin_amdgcn_tensor_load_to_lds) &&                     \
    __has_builtin(__builtin_amdgcn_s_wait_tensorcnt)
#define HAVE_TDM 1
#else
#define HAVE_TDM 0
#endif

// Device-pass-only probes: tell us in stderr which data-movement path compiled.
#if defined(__HIP_DEVICE_COMPILE__)
#if !HAVE_TDM
#warning "CDNA5 probe: TDM tensor_load_to_lds builtin NOT available (device pass)"
#endif
#if !HAVE_ASYNC_LDS
#warning "CDNA5 probe: global_load_async_to_lds builtin NOT available (device pass)"
#endif
#endif

__device__ __forceinline__ v8f wmma_bf16(v16bf a, v16bf b, v8f c) {
  // D = A(16x32 bf16) * B(32x16 bf16) + C(16x16 f32)
  return __builtin_amdgcn_wmma_f32_16x16x32_bf16(false, a, false, b, (short)0, c,
                                                 false, false);
}

__device__ __forceinline__ float sigm(float x) { return 1.f / (1.f + __expf(-x)); }

__device__ __forceinline__ unsigned lds_off32(const void* p) {
  return (unsigned)(size_t)(__attribute__((address_space(3))) const void*)p;
}

// TDM: DMA one contiguous 8 KB tile (4096 bf16, one row) global -> LDS.
// D# per CDNA5 ISA: count=1, type=2 (image),
// data_size=1 (2B), tensor_dim0=tile_dim0=4096, tile_dim1=1, stride=4096.
__device__ __forceinline__ void tdm_tile_load(const bf16* gsrc, unsigned ldsoff) {
#if HAVE_TDM
  unsigned long long ga = (unsigned long long)(size_t)gsrc;
  v4i_ g0;
  g0[0] = 1;                                             // count=1, user D#
  g0[1] = (int)ldsoff;                                   // lds_addr (bytes)
  g0[2] = (int)(unsigned)(ga & 0xffffffffull);           // global_addr[31:0]
  g0[3] = (int)(((unsigned)(ga >> 32) & 0x1ffffffu) | 0x80000000u); // addr[56:32] | type=2
  v8i_ g1;
  g1[0] = 0x00010000;                 // data_size=1 (2 bytes); no multicast/pad
  g1[1] = (int)(4096u << 16);         // tensor_dim0[15:0] << 16
  g1[2] = (int)(1u << 16);            // tensor_dim0[31:16]=0 | tensor_dim1[15:0]=1
  g1[3] = (int)(4096u << 16);         // tensor_dim1 hi=0 | tile_dim0=4096
  g1[4] = 1;                          // tile_dim1=1, tile_dim2=0
  g1[5] = 4096;                       // tensor_dim0_stride[31:0]
  g1[6] = (int)(4096u << 16);         // stride hi=0 | tensor_dim1_stride lo
  g1[7] = 0;
  v4i_ z4 = {0, 0, 0, 0};
  v8i_ z8 = {0, 0, 0, 0, 0, 0, 0, 0};
  __builtin_amdgcn_tensor_load_to_lds(g0, g1, z4, z4, z8, 0);
#else
  (void)gsrc; (void)ldsoff;
#endif
}

// Per-lane copy of one 16x256 bf16 tile (8 KB) global -> LDS with 256 threads.
// Async path: 2 x GLOBAL_LOAD_ASYNC_TO_LDS_B128 per thread (ASYNCcnt += 2/wave).
__device__ __forceinline__ void copy_tile(const bf16* __restrict__ gsrc,
                                          bf16* __restrict__ ldst, int tidx) {
#if HAVE_ASYNC_LDS
  for (int i = tidx; i < 512; i += 256) {
    __builtin_amdgcn_global_load_async_to_lds_b128(
        (__attribute__((address_space(1))) v4i_*)(const_cast<bf16*>(gsrc) + i * 8),
        (__attribute__((address_space(3))) v4i_*)(ldst + i * 8), 0, 0);
  }
#else
  for (int i = tidx; i < 512; i += 256) {
    v8bf tmp = *(const v8bf*)(gsrc + i * 8);
    *(v8bf*)(ldst + i * 8) = tmp;
  }
#endif
}

// ---------------------------------------------------------------------------
// Kernel A: build time-major bf16 x = q_emb[q]+a_emb[a]; convert W_ih / W_hh
// ---------------------------------------------------------------------------
__global__ __launch_bounds__(256) void prep_kernel(
    const int* __restrict__ questions, const int* __restrict__ answers,
    const float* __restrict__ q_emb, const float* __restrict__ a_emb,
    const float* __restrict__ w_ih, const float* __restrict__ w_hh,
    bf16* __restrict__ x_bf, bf16* __restrict__ wih_bf, bf16* __restrict__ whh_bf) {
  long long idx = (long long)blockIdx.x * 256 + threadIdx.x;
  if (idx < NX) {
    long long row = idx >> 8;           // time-major: row = t*64 + b
    int e = (int)(idx & 255);
    int t = (int)(row >> 6);
    int b = (int)(row & 63);
    int q = questions[b * T + t]; if (q < 0) q = 0;
    int a = answers[b * T + t];   if (a < 0) a = 0;
    float v = q_emb[(long long)q * H + e] + a_emb[(long long)a * H + e];
    x_bf[idx] = (bf16)v;
  } else if (idx < NX + NW) {
    long long i = idx - NX;
    wih_bf[i] = (bf16)w_ih[i];
  } else if (idx < NX + 2 * NW) {
    long long i = idx - NX - NW;
    whh_bf[i] = (bf16)w_hh[i];
  }
}

// ---------------------------------------------------------------------------
// Kernel B: gi[12800 x 768] = x_bf (12800x256) * W_ihT + b_ih   (bf16 WMMA)
// A register-resident per wave (full K); B tiles double-buffered in LDS.
// Prologue tile arrives via per-lane async global->LDS copies (ASYNCcnt);
// steady-state tiles are prefetched by the Tensor Data Mover (TENSORcnt,
// issued by wave 0). Falls back gracefully if either builtin is missing.
// ---------------------------------------------------------------------------
__global__ __launch_bounds__(256) void gemm_gi(
    const bf16* __restrict__ x, const bf16* __restrict__ wih,
    const float* __restrict__ b_ih, float* __restrict__ gi) {
  __shared__ __align__(32) bf16 btile[2][16 * H];   // 2 x 8 KB

  const int tid  = threadIdx.x;
  const int wave = tid >> 5;
  const int lane = tid & 31;
  const int half = lane >> 4;
  const int ln   = lane & 15;
  const int mt   = blockIdx.x * 8 + wave;    // 0..799 row tile

  // A fragments for the full K=256 of this wave's 16-row strip (64 VGPRs)
  ABfrag afr[8];
  const bf16* arow = x + ((size_t)(mt * 16 + ln)) * H + half * 16;
  for (int ks = 0; ks < 8; ++ks) {
    afr[ks].h[0] = *(const v8bf*)(arow + ks * 32);
    afr[ks].h[1] = *(const v8bf*)(arow + ks * 32 + 8);
  }

  // prologue: first B tile via async global->LDS copy (all waves participate)
  copy_tile(wih, btile[0], tid);

  for (int nt = 0; nt < 48; ++nt) {
    const int cur = nt & 1;
#if HAVE_TDM
    if (wave == 0) {
      if (nt < 47) {
        tdm_tile_load(wih + (size_t)(nt + 1) * 16 * H, lds_off32(&btile[cur ^ 1][0]));
        __builtin_amdgcn_s_wait_tensorcnt(1);   // TDM tile `nt` landed
      } else {
        __builtin_amdgcn_s_wait_tensorcnt(0);
      }
    }
#if HAVE_ASYNC_LDS
    __builtin_amdgcn_s_wait_asynccnt(0);        // drain prologue copy (no-op later)
#endif
#else
    if (nt < 47) copy_tile(wih + (size_t)(nt + 1) * 16 * H, btile[cur ^ 1], tid);
#if HAVE_ASYNC_LDS
    if (nt < 47) __builtin_amdgcn_s_wait_asynccnt(2);   // tile `nt` landed
    else         __builtin_amdgcn_s_wait_asynccnt(0);
#endif
#endif
    __syncthreads();

    // B fragments from LDS feeding the 8-WMMA accumulate chain
    ABfrag bfr[8];
    const bf16* bbase = &btile[cur][ln * H + half * 16];
    for (int ks = 0; ks < 8; ++ks)
      bfr[ks].v = *(const v16bf*)(bbase + ks * 32);
    v8f acc = {0.f, 0.f, 0.f, 0.f, 0.f, 0.f, 0.f, 0.f};
    for (int ks = 0; ks < 8; ++ks)
      acc = wmma_bf16(afr[ks].v, bfr[ks].v, acc);

    const int col = nt * 16 + ln;
    const float bias = b_ih[col];
    for (int j = 0; j < 8; ++j) {
      int row = mt * 16 + j + 8 * half;      // C layout: VGPR j -> M=j+8*half
      gi[(size_t)row * G3 + col] = acc[j] + bias;
    }
    __syncthreads();   // done reading btile[cur] before it is overwritten
  }
}

// ---------------------------------------------------------------------------
// Kernel C: persistent single-workgroup GRU scan (32 waves = 1024 threads).
// h held fp32 in registers (C-fragment layout), bf16 mirror in LDS for WMMA A.
// Wave w owns triplets (mt,nt) for tri = 2w, 2w+1:  mt=tri>>4, nt=tri&15.
// Gate tiles for columns nt*16.. use r/z/n at col, col+256, col+512.
// ---------------------------------------------------------------------------
__global__ __launch_bounds__(1024) void gru_scan(
    const bf16* __restrict__ whh, const float* __restrict__ gi,
    const float* __restrict__ b_hh, const int* __restrict__ questions,
    const float* __restrict__ pred_w, const float* __restrict__ pred_b,
    float* __restrict__ out) {
  __shared__ __align__(32) bf16 h_s[Bsz * H];   // 32 KB
  __shared__ float pred_acc[Bsz];

  const int tid  = threadIdx.x;
  const int wave = tid >> 5;
  const int lane = tid & 31;
  const int half = lane >> 4;
  const int ln   = lane & 15;

  for (int i = tid; i < Bsz * H; i += 1024) h_s[i] = (bf16)0.f;
  if (tid < Bsz) pred_acc[tid] = 0.f;

  v8f hreg[2];
  hreg[0] = {0.f, 0.f, 0.f, 0.f, 0.f, 0.f, 0.f, 0.f};
  hreg[1] = hreg[0];
  __syncthreads();

  for (int t = 0; t < T; ++t) {
    // warm WGP$/L2 with next step's gi slice while the matrix pipe works
    if (t + 1 < T) {
      const float* gnext = gi + (size_t)(t + 1) * Bsz * G3;
      for (int i = tid * 32; i < Bsz * G3; i += 1024 * 32)
        __builtin_prefetch(gnext + i, 0, 1);
    }

    v8f accr[2], accz[2], accn[2];
    // ---- phase 1: gh = h @ W_hh^T via WMMA (reads h_s, streams W_hh from L2)
    for (int trip = 0; trip < 2; ++trip) {
      const int tri = wave * 2 + trip;
      const int mt = tri >> 4, nt = tri & 15;
      const int colr = nt * 16 + ln;
      const bf16* br = whh + (size_t)colr * H;
      const bf16* bz = whh + (size_t)(colr + 256) * H;
      const bf16* bn = whh + (size_t)(colr + 512) * H;
      v8f cr = {0.f,0.f,0.f,0.f,0.f,0.f,0.f,0.f};
      v8f cz = cr, cn = cr;
      const bf16* ar = &h_s[(mt * 16 + ln) * H + half * 16];
      for (int ks = 0; ks < 8; ++ks) {
        const int k0 = ks * 32;
        ABfrag a;
        a.h[0] = *(const v8bf*)(ar + k0);
        a.h[1] = *(const v8bf*)(ar + k0 + 8);
        const int bo = k0 + half * 16;
        cr = wmma_bf16(a.v, *(const v16bf*)(br + bo), cr);
        cz = wmma_bf16(a.v, *(const v16bf*)(bz + bo), cz);
        cn = wmma_bf16(a.v, *(const v16bf*)(bn + bo), cn);
      }
      accr[trip] = cr; accz[trip] = cz; accn[trip] = cn;
    }
    __syncthreads();   // everyone done reading old h_s

    // ---- phase 2: gate math in registers, write new h, prediction partials
    for (int trip = 0; trip < 2; ++trip) {
      const int tri = wave * 2 + trip;
      const int mt = tri >> 4, nt = tri & 15;
      const int c = nt * 16 + ln;
      const float bhr = b_hh[c], bhz = b_hh[c + 256], bhn = b_hh[c + 512];
      for (int j = 0; j < 8; ++j) {
        const int b = mt * 16 + j + 8 * half;
        const float* girow = gi + ((size_t)t * Bsz + b) * G3;
        const float ir = girow[c], iz = girow[c + 256], in_ = girow[c + 512];
        const float ghr = accr[trip][j] + bhr;
        const float ghz = accz[trip][j] + bhz;
        const float ghn = accn[trip][j] + bhn;
        const float r = sigm(ir + ghr);
        const float z = sigm(iz + ghz);
        const float pre = in_ + r * ghn;
        const float e2 = __expf(2.f * pre);
        const float n = (e2 - 1.f) / (e2 + 1.f);
        const float hnew = (1.f - z) * n + z * hreg[trip][j];
        hreg[trip][j] = hnew;
        h_s[b * H + c] = (bf16)hnew;
        // prediction partial: dot(h_new[b,:], pred_w[q_t[b],:]) on this slice
        int q = questions[b * T + t]; if (q < 0) q = 0;
        float p = hnew * pred_w[(size_t)q * H + c];
        p += __shfl_xor(p, 1, 32);
        p += __shfl_xor(p, 2, 32);
        p += __shfl_xor(p, 4, 32);
        p += __shfl_xor(p, 8, 32);
        if (ln == 0) atomicAdd(&pred_acc[b], p);   // ds_add_f32
      }
    }
    __syncthreads();

    // ---- phase 3: finalize probability, reset accumulators
    if (tid < Bsz) {
      int q = questions[tid * T + t]; if (q < 0) q = 0;
      float prob = sigm(pred_acc[tid] + pred_b[q]);
      if (t < T - 1) out[(size_t)tid * (T - 1) + t] = prob;
      pred_acc[tid] = 0.f;
    }
    __syncthreads();
  }
}

// ---------------------------------------------------------------------------
extern "C" void kernel_launch(void* const* d_in, const int* in_sizes, int n_in,
                              void* d_out, int out_size, void* d_ws, size_t ws_size,
                              hipStream_t stream) {
  (void)in_sizes; (void)n_in; (void)out_size; (void)ws_size;
  // setup_inputs() order
  const int*   questions = (const int*)d_in[1];
  const int*   answers   = (const int*)d_in[2];
  const float* q_emb     = (const float*)d_in[3];
  const float* a_emb     = (const float*)d_in[4];
  const float* w_ih      = (const float*)d_in[5];
  const float* w_hh      = (const float*)d_in[6];
  const float* b_ih      = (const float*)d_in[7];
  const float* b_hh      = (const float*)d_in[8];
  const float* pred_w    = (const float*)d_in[9];
  const float* pred_b    = (const float*)d_in[10];
  float* out = (float*)d_out;

  // workspace layout (bytes)
  char* ws = (char*)d_ws;
  bf16* x_bf   = (bf16*)(ws);                                  // 12800*256*2 = 6,553,600
  bf16* wih_bf = (bf16*)(ws + 6553600);                        // 393,216
  bf16* whh_bf = (bf16*)(ws + 6553600 + 393216);               // 393,216
  float* gi    = (float*)(ws + 6553600 + 2 * 393216);          // 12800*768*4 = 39,321,600

  // A: gathers + bf16 conversion
  {
    long long total = NX + 2 * NW;
    int blocks = (int)((total + 255) / 256);
    prep_kernel<<<blocks, 256, 0, stream>>>(questions, answers, q_emb, a_emb,
                                            w_ih, w_hh, x_bf, wih_bf, whh_bf);
  }
  // B: input-side GEMM, 100 blocks x 8 waves = 800 row tiles; nt loop inside
  gemm_gi<<<100, 256, 0, stream>>>(x_bf, wih_bf, b_ih, gi);
  // C: persistent sequential scan, one workgroup of 32 waves
  gru_scan<<<1, 1024, 0, stream>>>(whh_bf, gi, b_hh, questions, pred_w, pred_b, out);
}
